// PhiRhoLinear_24704651886628
// MI455X (gfx1250) — matrix operand order repaired and checked by reading, not compiled, not run on hardware
//
#include <hip/hip_runtime.h>
#include <hip/hip_bf16.h>

typedef __attribute__((ext_vector_type(16))) _Float16 v16h;
typedef __attribute__((ext_vector_type(8)))  _Float16 v8h;
typedef __attribute__((ext_vector_type(8)))  float    v8f;
typedef __attribute__((ext_vector_type(4)))  int      vi4;
typedef __attribute__((ext_vector_type(4)))  float    vf4;

// pointer types for the async global->LDS builtin (signature revealed by clang:
// params are int __vector(4) pointers in AS(1)/AS(3))
typedef __attribute__((address_space(1))) vi4 vi4_g;
typedef __attribute__((address_space(3))) vi4 vi4_l;

static constexpr int K_DIM = 4096;
static constexpr int N_DIM = 11008;
static constexpr int M_DIM = 8192;

static constexpr int BM  = 128;
static constexpr int BN  = 128;
static constexpr int BK  = 32;       // one WMMA K-step per stage
static constexpr int LDH = BK + 8;   // padded LDS row (halfs): conflict-free b128 reads

// ---- CDNA5 async global->LDS copy (guarded: falls back to VGPR staging) ----
#if defined(__has_builtin)
#  if __has_builtin(__builtin_amdgcn_global_load_async_to_lds_b128)
#    define HAS_ASYNC_COPY 1
#  else
#    define HAS_ASYNC_COPY 0
#  endif
#  if __has_builtin(__builtin_amdgcn_s_wait_asynccnt)
#    define WAIT_ASYNC0() __builtin_amdgcn_s_wait_asynccnt(0)
#  else
#    define WAIT_ASYNC0() asm volatile("s_wait_asynccnt 0x0" ::: "memory")
#  endif
#else
#  define HAS_ASYNC_COPY 0
#  define WAIT_ASYNC0()
#endif

// ---------------------------------------------------------------------------
// Pre-pass 1: dequantize weights  W_f16[o,k] = (f16) codebook[idx[o,k]]
// Reads the 180MB int32 index array exactly once; writes 90MB f16 W
// (which then lives in the 192MB L2 for the whole GEMM).
// ---------------------------------------------------------------------------
__global__ __launch_bounds__(256)
void dequant_w(const int* __restrict__ idx,
               const float* __restrict__ cb_g,
               _Float16* __restrict__ wq) {
  __shared__ float cb[256];
  cb[threadIdx.x] = cb_g[threadIdx.x];       // blockDim == 256 == NUM_CODES
  __syncthreads();
  const size_t base = ((size_t)blockIdx.x * 256 + threadIdx.x) * 8;
  vi4 i0 = *(const vi4*)(idx + base);
  vi4 i1 = *(const vi4*)(idx + base + 4);
  v8h o;
#pragma unroll
  for (int j = 0; j < 4; ++j) o[j]     = (_Float16)cb[i0[j]];
#pragma unroll
  for (int j = 0; j < 4; ++j) o[4 + j] = (_Float16)cb[i1[j]];
  *(v8h*)(wq + base) = o;
}

// ---------------------------------------------------------------------------
// Pre-pass 2: convert x (f32 -> f16)
// ---------------------------------------------------------------------------
__global__ __launch_bounds__(256)
void cvt_x(const float* __restrict__ x, _Float16* __restrict__ xh) {
  const size_t base = ((size_t)blockIdx.x * 256 + threadIdx.x) * 8;
  vf4 a = *(const vf4*)(x + base);
  vf4 b = *(const vf4*)(x + base + 4);
  v8h o;
#pragma unroll
  for (int j = 0; j < 4; ++j) o[j]     = (_Float16)a[j];
#pragma unroll
  for (int j = 0; j < 4; ++j) o[4 + j] = (_Float16)b[j];
  *(v8h*)(xh + base) = o;
}

// ---------------------------------------------------------------------------
// WMMA GEMM: C[m,n] = sum_k A[m,k] * W[n,k] + bias[n]
// 256 threads = 8 waves; block tile 128x128; waves 4(M) x 2(N);
// each wave: 2x4 grid of 16x16 f32 accumulators, v_wmma_f32_16x16x32_f16.
// Double-buffered LDS (40KB), staged with ASYNC global->LDS copies when
// available (ASYNCcnt), else VGPR-staged b128 loads + ds_store_b128.
// ---------------------------------------------------------------------------
__global__ __launch_bounds__(256)
void wmma_gemm(const _Float16* __restrict__ A,   // [M, K] row-major (x in f16)
               const _Float16* __restrict__ W,   // [N, K] row-major (dequant W)
               const float* __restrict__ bias,   // [N]
               float* __restrict__ C) {          // [M, N]
  __shared__ _Float16 sA[2][BM * LDH];
  __shared__ _Float16 sB[2][BN * LDH];

  const int tid  = threadIdx.x;
  const int lane = tid & 31;
  const int l16  = lane & 15;
  const bool hiL = (lane >= 16);
  const int wid  = tid >> 5;
  const int wm   = wid & 3;    // 4 waves along M (32 rows each)
  const int wn   = wid >> 2;   // 2 waves along N (64 cols each)

  const size_t Mbase = (size_t)blockIdx.y * BM;
  const size_t Nbase = (size_t)blockIdx.x * BN;

  const _Float16* gA = A + Mbase * K_DIM;
  const _Float16* gB = W + Nbase * K_DIM;

  // cooperative staging: thread covers (row = tid>>2 [+64]), 16B chunk (tid&3)
  const int srow   = tid >> 2;         // 0..63
  const int schunk = (tid & 3) * 8;    // half offset 0/8/16/24

  v8f acc[2][4] = {};

  const int NS = K_DIM / BK;           // 128 stages

#if HAS_ASYNC_COPY
  // ---------------- async global->LDS staging (ASYNCcnt) ----------------
  {
#pragma unroll
    for (int r = 0; r < 2; ++r) {
      const int row = srow + r * 64;
      __builtin_amdgcn_global_load_async_to_lds_b128(
          (vi4_g*)(gA + (size_t)row * K_DIM + schunk),
          (vi4_l*)(&sA[0][row * LDH + schunk]), 0, 0);
      __builtin_amdgcn_global_load_async_to_lds_b128(
          (vi4_g*)(gB + (size_t)row * K_DIM + schunk),
          (vi4_l*)(&sB[0][row * LDH + schunk]), 0, 0);
    }
  }
  WAIT_ASYNC0();
  __syncthreads();
#else
  // ---------------- fallback: VGPR-staged loads ----------------
  vi4 ra[2], rb[2];
#pragma unroll
  for (int r = 0; r < 2; ++r) {
    const size_t row = (size_t)(srow + r * 64);
    ra[r] = *(const vi4*)(gA + row * K_DIM + schunk);
    rb[r] = *(const vi4*)(gB + row * K_DIM + schunk);
  }
#pragma unroll
  for (int r = 0; r < 2; ++r) {
    const int row = srow + r * 64;
    *(vi4*)(&sA[0][row * LDH + schunk]) = ra[r];
    *(vi4*)(&sB[0][row * LDH + schunk]) = rb[r];
  }
  __syncthreads();
#endif

  for (int s = 0; s < NS; ++s) {
    const int cur = s & 1;
    const int nxt = s + 1;

    if (nxt < NS) {
#if HAS_ASYNC_COPY
#pragma unroll
      for (int r = 0; r < 2; ++r) {
        const int row = srow + r * 64;
        __builtin_amdgcn_global_load_async_to_lds_b128(
            (vi4_g*)(gA + (size_t)row * K_DIM + (size_t)nxt * BK + schunk),
            (vi4_l*)(&sA[nxt & 1][row * LDH + schunk]), 0, 0);
        __builtin_amdgcn_global_load_async_to_lds_b128(
            (vi4_g*)(gB + (size_t)row * K_DIM + (size_t)nxt * BK + schunk),
            (vi4_l*)(&sB[nxt & 1][row * LDH + schunk]), 0, 0);
      }
#else
#pragma unroll
      for (int r = 0; r < 2; ++r) {
        const size_t row = (size_t)(srow + r * 64);
        ra[r] = *(const vi4*)(gA + row * K_DIM + (size_t)nxt * BK + schunk);
        rb[r] = *(const vi4*)(gB + row * K_DIM + (size_t)nxt * BK + schunk);
      }
#endif
      if (nxt + 3 < NS) {   // deep prefetch -> global_prefetch_b8
        __builtin_prefetch(gA + (size_t)srow * K_DIM + (size_t)(nxt + 3) * BK + schunk, 0, 1);
        __builtin_prefetch(gB + (size_t)srow * K_DIM + (size_t)(nxt + 3) * BK + schunk, 0, 1);
      }
    }

    // --- A fragments: lane = row M; lane-group 0: K 0-7,16-23; group 1: K 8-15,24-31
    v16h af[2];
#pragma unroll
    for (int mt = 0; mt < 2; ++mt) {
      const int row  = wm * 32 + mt * 16 + l16;
      const int base = row * LDH + (hiL ? 8 : 0);
      v8h lo = *(const v8h*)(&sA[cur][base]);
      v8h hh = *(const v8h*)(&sA[cur][base + 16]);
      af[mt] = __builtin_shufflevector(lo, hh, 0,1,2,3,4,5,6,7,8,9,10,11,12,13,14,15);
    }
    // --- B fragments: lane = col N (= row of W); group 0: K 0-7,16-23; group 1: K 8-15,24-31
    v16h bf[4];
#pragma unroll
    for (int nt = 0; nt < 4; ++nt) {
      const int row  = wn * 64 + nt * 16 + l16;
      const int base = row * LDH + (hiL ? 8 : 0);
      v8h lo = *(const v8h*)(&sB[cur][base]);
      v8h hh = *(const v8h*)(&sB[cur][base + 16]);
      bf[nt] = __builtin_shufflevector(lo, hh, 0,1,2,3,4,5,6,7,8,9,10,11,12,13,14,15);
    }

#pragma unroll
    for (int mt = 0; mt < 2; ++mt)
#pragma unroll
      for (int nt = 0; nt < 4; ++nt)
        acc[mt][nt] = __builtin_amdgcn_wmma_f32_16x16x32_f16(
            false, af[mt], false, bf[nt], (short)0, acc[mt][nt], false, false);

    if (nxt < NS) {
#if HAS_ASYNC_COPY
      WAIT_ASYNC0();     // my stage-(s+1) tiles are in LDS before the barrier
#else
#pragma unroll
      for (int r = 0; r < 2; ++r) {
        const int row = srow + r * 64;
        *(vi4*)(&sA[nxt & 1][row * LDH + schunk]) = ra[r];
        *(vi4*)(&sB[nxt & 1][row * LDH + schunk]) = rb[r];
      }
#endif
    }
    __syncthreads();
  }

  // Epilogue: C/D layout — lane<16: M=v, N=lane; lane>=16: M=v+8, N=lane-16
#pragma unroll
  for (int mt = 0; mt < 2; ++mt) {
    const size_t rbase = Mbase + (size_t)(wm * 32 + mt * 16 + (hiL ? 8 : 0));
#pragma unroll
    for (int nt = 0; nt < 4; ++nt) {
      const size_t col = Nbase + (size_t)(wn * 64 + nt * 16 + l16);
      const float bv = bias[col];
      float* cp = C + rbase * N_DIM + col;
#pragma unroll
      for (int v = 0; v < 8; ++v)
        cp[(size_t)v * N_DIM] = acc[mt][nt][v] + bv;
    }
  }
}

// ---------------------------------------------------------------------------
extern "C" void kernel_launch(void* const* d_in, const int* in_sizes, int n_in,
                              void* d_out, int out_size, void* d_ws, size_t ws_size,
                              hipStream_t stream) {
  (void)in_sizes; (void)n_in; (void)out_size; (void)ws_size;

  const float* x    = (const float*)d_in[0];   // [4,2048,4096] f32
  const int*   widx = (const int*)  d_in[1];   // [11008,4096] i32
  const float* cb   = (const float*)d_in[2];   // [256] f32
  const float* bias = (const float*)d_in[3];   // [11008] f32
  float*       out  = (float*)d_out;           // [8192,11008] f32

  _Float16* wq = (_Float16*)d_ws;                       // 90MB  f16 W
  _Float16* xh = wq + (size_t)N_DIM * K_DIM;            // 67MB  f16 x

  const unsigned dq_blocks = (unsigned)(((size_t)N_DIM * K_DIM) / (8 * 256)); // 22016
  const unsigned cx_blocks = (unsigned)(((size_t)M_DIM * K_DIM) / (8 * 256)); // 16384

  dequant_w<<<dq_blocks, 256, 0, stream>>>(widx, cb, wq);
  cvt_x   <<<cx_blocks, 256, 0, stream>>>(x, xh);
  wmma_gemm<<<dim3(N_DIM / BN, M_DIM / BM), 256, 0, stream>>>(xh, wq, bias, out);
}
